// MultiHeadAttention_3556232921482
// MI455X (gfx1250) — matrix operand order repaired and checked by reading, compiled, not run on hardware
//
#include <hip/hip_runtime.h>
#include <stdint.h>
#include <math.h>

#define B_N   4
#define SEQ   2048
#define DM    1024
#define NH    16
#define HD    64
#define NROWS (B_N * SEQ)   // 8192

typedef __attribute__((ext_vector_type(16))) __bf16 v16bf;
typedef __attribute__((ext_vector_type(8)))  __bf16 v8bf;
typedef __attribute__((ext_vector_type(8)))  float  v8f;
typedef __attribute__((ext_vector_type(4)))  unsigned int u32x4;
typedef __attribute__((ext_vector_type(4)))  float  f32x4;

__device__ __forceinline__ unsigned short f2bf(float f) {
    unsigned int u = __builtin_bit_cast(unsigned int, f);
    u += 0x7FFFu + ((u >> 16) & 1u);     // round-to-nearest-even
    return (unsigned short)(u >> 16);
}
__device__ __forceinline__ unsigned int pack2(float lo, float hi) {
    return (unsigned int)f2bf(lo) | ((unsigned int)f2bf(hi) << 16);
}
__device__ __forceinline__ v16bf frag_cat(u32x4 lo, u32x4 hi) {
    v8bf a = __builtin_bit_cast(v8bf, lo);
    v8bf b = __builtin_bit_cast(v8bf, hi);
    return __builtin_shufflevector(a, b, 0,1,2,3,4,5,6,7,8,9,10,11,12,13,14,15);
}

// A fragment (16 rows x 32 K, bf16, LDS row stride ld elems).
// lanes 0-15 rows 0-15 K 0..15; lanes 16-31 rows 0-15 K 16..31 (contiguous, matches B)
__device__ __forceinline__ v16bf load_fragA(const unsigned short* p, int ld, int lane) {
    int row = lane & 15;
    int kg  = (lane >> 4) & 1;
    const unsigned short* r = p + row * ld;
    u32x4 lo = *(const u32x4*)(r + kg * 16);
    u32x4 hi = *(const u32x4*)(r + kg * 16 + 8);
    return frag_cat(lo, hi);
}
// B fragment from transposed storage Bt[col][k] (16 cols x 32 K):
// lane holds column (lane&15); lanes 0-15 K 0..15, lanes 16-31 K 16..31 (contiguous)
__device__ __forceinline__ v16bf load_fragB(const unsigned short* p, int ld, int lane) {
    int col = lane & 15;
    int kh  = (lane >> 4) & 1;
    const unsigned short* r = p + col * ld + kh * 16;
    u32x4 lo = *(const u32x4*)(r);
    u32x4 hi = *(const u32x4*)(r + 8);
    return frag_cat(lo, hi);
}

__global__ void f32_to_bf16_kernel(const float* __restrict__ in,
                                   unsigned short* __restrict__ out, int n) {
    int idx = (blockIdx.x * 256 + threadIdx.x) * 8;
    if (idx + 7 < n) {
        f32x4 a = *(const f32x4*)(in + idx);
        f32x4 b = *(const f32x4*)(in + idx + 4);
        u32x4 w;
        w.x = pack2(a.x, a.y); w.y = pack2(a.z, a.w);
        w.z = pack2(b.x, b.y); w.w = pack2(b.z, b.w);
        *(u32x4*)(out + idx) = w;
    }
}

// W f32 [K][N] -> Wt bf16 [N][K]
__global__ __launch_bounds__(256)
void transpose_w_kernel(const float* __restrict__ W, unsigned short* __restrict__ Wt) {
    __shared__ float tile[32][33];
    int bx = blockIdx.x * 32;     // n base
    int by = blockIdx.y * 32;     // k base
    int tx = threadIdx.x & 31;
    int ty = threadIdx.x >> 5;    // 0..7
    for (int i = 0; i < 32; i += 8)
        tile[ty + i][tx] = W[(size_t)(by + ty + i) * DM + bx + tx];
    __syncthreads();
    for (int i = 0; i < 32; i += 8)
        Wt[(size_t)(bx + ty + i) * DM + by + tx] = f2bf(tile[tx][ty + i]);
}

enum { MODE_QK = 0, MODE_V = 1, MODE_OUT = 2, MODE_LOGITS = 3 };

// C = A(bf16 [M,K], ld=lda) x Bt(bf16 [N,K], ld=ldb)^T, double-buffered LDS pipeline.
template <int MODE>
__global__ __launch_bounds__(256)
void gemm_bf16_kernel(const unsigned short* __restrict__ A, int lda,
                      const unsigned short* __restrict__ Bt, int ldb,
                      const float* __restrict__ bias,
                      const float* __restrict__ mask, float scale,
                      void* __restrict__ outp, int Kdim) {
    constexpr int BM = 128, BN = 128, BK = 32;
    constexpr int LDSS = BK + 8;                 // pad to spread LDS banks
    constexpr int BUFO = BM * LDSS;              // elems per buffer (sA); sB same
    __shared__ unsigned short sA[2][BUFO];
    __shared__ unsigned short sB[2][BUFO];

    int tid  = threadIdx.x;
    int lane = tid & 31;
    int wave = tid >> 5;       // 8 waves
    int wm   = wave >> 2;      // 0..1 -> 64-row strip
    int wn   = wave & 3;       // 0..3 -> 32-col strip
    int m0   = blockIdx.x * BM;
    int n0   = blockIdx.y * BN;
    int bh   = blockIdx.z;

    const unsigned short* Ap = A;
    const unsigned short* Bp = Bt;
    if (MODE == MODE_LOGITS) {
        Ap = A  + (size_t)bh * SEQ * HD;
        Bp = Bt + (size_t)bh * SEQ * HD;
    }

    const v8f vzero = {0.f,0.f,0.f,0.f,0.f,0.f,0.f,0.f};
    v8f c[4][2];
    for (int i = 0; i < 4; i++)
        for (int j = 0; j < 2; j++) c[i][j] = vzero;

    int lrow = tid >> 1;           // 0..127
    int lcol = (tid & 1) * 16;     // 0 / 16
    const unsigned short* gA = Ap + (size_t)(m0 + lrow) * lda + lcol;
    const unsigned short* gB = Bp + (size_t)(n0 + lrow) * ldb + lcol;
    unsigned short* swA = &sA[0][0] + lrow * LDSS + lcol;
    unsigned short* swB = &sB[0][0] + lrow * LDSS + lcol;

    // prologue: stage k-step 0 into buffer 0
    {
        u32x4 a0 = *(const u32x4*)gA;
        u32x4 a1 = *(const u32x4*)(gA + 8);
        u32x4 b0 = *(const u32x4*)gB;
        u32x4 b1 = *(const u32x4*)(gB + 8);
        *(u32x4*)swA = a0; *(u32x4*)(swA + 8) = a1;
        *(u32x4*)swB = b0; *(u32x4*)(swB + 8) = b1;
    }
    __syncthreads();

    int nk = Kdim / BK;
#pragma unroll 1
    for (int kt = 0; kt < nk - 1; kt++) {        // steady state: always prefetch
        int cur = kt & 1;
        if (kt + 4 < nk) {                       // L2 prefetch, distance 4 (uniform branch)
            __builtin_prefetch(gA + (kt + 4) * BK, 0, 3);
            __builtin_prefetch(gB + (kt + 4) * BK, 0, 3);
        }
        const unsigned short* pa = gA + (kt + 1) * BK;
        const unsigned short* pb = gB + (kt + 1) * BK;
        u32x4 ra0 = *(const u32x4*)pa;
        u32x4 ra1 = *(const u32x4*)(pa + 8);
        u32x4 rb0 = *(const u32x4*)pb;
        u32x4 rb1 = *(const u32x4*)(pb + 8);

        const unsigned short* cA = &sA[0][0] + cur * BUFO;
        const unsigned short* cB = &sB[0][0] + cur * BUFO;
        v16bf af[4], bf2[2];
        for (int i = 0; i < 4; i++)
            af[i] = load_fragA(cA + (wm * 64 + i * 16) * LDSS, LDSS, lane);
        for (int j = 0; j < 2; j++)
            bf2[j] = load_fragB(cB + (wn * 32 + j * 16) * LDSS, LDSS, lane);
        for (int i = 0; i < 4; i++)
            for (int j = 0; j < 2; j++)
                c[i][j] = __builtin_amdgcn_wmma_f32_16x16x32_bf16(
                    false, af[i], false, bf2[j], (short)0, c[i][j], false, false);

        unsigned short* dA = swA + (cur ^ 1) * BUFO;
        unsigned short* dB = swB + (cur ^ 1) * BUFO;
        *(u32x4*)dA = ra0; *(u32x4*)(dA + 8) = ra1;
        *(u32x4*)dB = rb0; *(u32x4*)(dB + 8) = rb1;
        __syncthreads();
    }
    {   // final k-step (no prefetch, no staging)
        int cur = (nk - 1) & 1;
        const unsigned short* cA = &sA[0][0] + cur * BUFO;
        const unsigned short* cB = &sB[0][0] + cur * BUFO;
        v16bf af[4], bf2[2];
        for (int i = 0; i < 4; i++)
            af[i] = load_fragA(cA + (wm * 64 + i * 16) * LDSS, LDSS, lane);
        for (int j = 0; j < 2; j++)
            bf2[j] = load_fragB(cB + (wn * 32 + j * 16) * LDSS, LDSS, lane);
        for (int i = 0; i < 4; i++)
            for (int j = 0; j < 2; j++)
                c[i][j] = __builtin_amdgcn_wmma_f32_16x16x32_bf16(
                    false, af[i], false, bf2[j], (short)0, c[i][j], false, false);
    }

    int nloc = lane & 15;
    int moff = (lane >> 4) * 8;
    for (int i = 0; i < 4; i++) {
        for (int j = 0; j < 2; j++) {
            int col = n0 + wn * 32 + j * 16 + nloc;
            float bval = 0.f;
            if (MODE != MODE_LOGITS) bval = bias[col];
            for (int r = 0; r < 8; r++) {
                int row = m0 + wm * 64 + i * 16 + moff + r;
                float val = c[i][j][r] + bval;
                if (MODE == MODE_QK) {
                    int b = row / SEQ, s = row % SEQ;
                    int h = col / HD,  d = col % HD;
                    ((unsigned short*)outp)[((size_t)(b * NH + h) * SEQ + s) * HD + d] = f2bf(val);
                } else if (MODE == MODE_V) {
                    int b = row / SEQ, s = row % SEQ;
                    int h = col / HD,  d = col % HD;
                    ((unsigned short*)outp)[((size_t)(b * NH + h) * HD + d) * SEQ + s] = f2bf(val);
                } else if (MODE == MODE_OUT) {
                    ((float*)outp)[(size_t)row * DM + col] = val;
                } else { // MODE_LOGITS: row = sq, col = sk; streaming store, bypass caches
                    int b = bh / NH;
                    float mv = mask[((size_t)b * SEQ + row) * SEQ + col];
                    val = val * scale + mv * -1e9f;
                    __builtin_nontemporal_store(
                        val, &((float*)outp)[((size_t)bh * SEQ + row) * SEQ + col]);
                }
            }
        }
    }
}

// rowwise softmax; reads/writes f32 in d_out (non-temporal) + emits bf16 copy for attn@V
__global__ __launch_bounds__(256)
void softmax_kernel(float* __restrict__ attn, unsigned short* __restrict__ attn_bf) {
    __shared__ float red[256];
    float*          p  = attn    + (size_t)blockIdx.x * SEQ;
    unsigned short* pb = attn_bf + (size_t)blockIdx.x * SEQ;
    int tid  = threadIdx.x;
    int base = tid * 8;

    f32x4 v0 = __builtin_nontemporal_load((const f32x4*)(p + base));
    f32x4 v1 = __builtin_nontemporal_load((const f32x4*)(p + base + 4));
    float m = fmaxf(fmaxf(fmaxf(v0.x, v0.y), fmaxf(v0.z, v0.w)),
                    fmaxf(fmaxf(v1.x, v1.y), fmaxf(v1.z, v1.w)));
    red[tid] = m; __syncthreads();
    for (int s = 128; s > 0; s >>= 1) {
        if (tid < s) red[tid] = fmaxf(red[tid], red[tid + s]);
        __syncthreads();
    }
    m = red[0]; __syncthreads();

    v0.x = __expf(v0.x - m); v0.y = __expf(v0.y - m);
    v0.z = __expf(v0.z - m); v0.w = __expf(v0.w - m);
    v1.x = __expf(v1.x - m); v1.y = __expf(v1.y - m);
    v1.z = __expf(v1.z - m); v1.w = __expf(v1.w - m);
    float sum = (v0.x + v0.y) + (v0.z + v0.w) + (v1.x + v1.y) + (v1.z + v1.w);
    red[tid] = sum; __syncthreads();
    for (int s = 128; s > 0; s >>= 1) {
        if (tid < s) red[tid] += red[tid + s];
        __syncthreads();
    }
    float inv = 1.f / red[0];
    v0 *= inv; v1 *= inv;

    __builtin_nontemporal_store(v0, (f32x4*)(p + base));
    __builtin_nontemporal_store(v1, (f32x4*)(p + base + 4));
    u32x4 w;
    w.x = pack2(v0.x, v0.y); w.y = pack2(v0.z, v0.w);
    w.z = pack2(v1.x, v1.y); w.w = pack2(v1.z, v1.w);
    __builtin_nontemporal_store(w, (u32x4*)(pb + base));
}

// ctx = attn_bf16([sq,sk]) x Vt(bf16 [64,sk])^T per (b,h); merged-heads bf16 out
__global__ __launch_bounds__(256)
void ctx_gemm_kernel(const unsigned short* __restrict__ attn_bf,
                     const unsigned short* __restrict__ Vt,
                     unsigned short* __restrict__ ctx) {
    constexpr int BM = 128, BK = 32, LDSS = BK + 8;
    constexpr int ABUF = BM * LDSS, BBUF = HD * LDSS;
    __shared__ unsigned short sA[2][ABUF];
    __shared__ unsigned short sB[2][BBUF];

    int tid  = threadIdx.x;
    int lane = tid & 31;
    int wave = tid >> 5;
    int wm   = wave >> 2;     // 0..1
    int wn   = wave & 3;      // 0..3 -> 16-col strip of 64
    int m0   = blockIdx.x * BM;
    int bh   = blockIdx.z;
    const unsigned short* Ap = attn_bf + (size_t)bh * SEQ * SEQ;
    const unsigned short* Bp = Vt      + (size_t)bh * HD * SEQ;

    const v8f vzero = {0.f,0.f,0.f,0.f,0.f,0.f,0.f,0.f};
    v8f c[4];
    for (int i = 0; i < 4; i++) c[i] = vzero;

    int arow = tid >> 1, acol = (tid & 1) * 16;
    int brow = tid >> 2, bcol = (tid & 3) * 8;
    const unsigned short* gA = Ap + (size_t)(m0 + arow) * SEQ + acol;
    const unsigned short* gB = Bp + (size_t)brow * SEQ + bcol;
    unsigned short* swA = &sA[0][0] + arow * LDSS + acol;
    unsigned short* swB = &sB[0][0] + brow * LDSS + bcol;

    {
        u32x4 a0 = __builtin_nontemporal_load((const u32x4*)gA);
        u32x4 a1 = __builtin_nontemporal_load((const u32x4*)(gA + 8));
        u32x4 b0 = *(const u32x4*)gB;
        *(u32x4*)swA = a0; *(u32x4*)(swA + 8) = a1;
        *(u32x4*)swB = b0;
    }
    __syncthreads();

    constexpr int nk = SEQ / BK;     // 64
#pragma unroll 1
    for (int kt = 0; kt < nk - 1; kt++) {
        int cur = kt & 1;
        if (kt + 4 < nk) {
            __builtin_prefetch(gA + (kt + 4) * BK, 0, 3);
            __builtin_prefetch(gB + (kt + 4) * BK, 0, 3);
        }
        const unsigned short* pa = gA + (kt + 1) * BK;
        u32x4 ra0 = __builtin_nontemporal_load((const u32x4*)pa);
        u32x4 ra1 = __builtin_nontemporal_load((const u32x4*)(pa + 8));
        u32x4 rb  = *(const u32x4*)(gB + (kt + 1) * BK);

        const unsigned short* cA = &sA[0][0] + cur * ABUF;
        const unsigned short* cB = &sB[0][0] + cur * BBUF;
        v16bf bfr = load_fragB(cB + (wn * 16) * LDSS, LDSS, lane);
        for (int i = 0; i < 4; i++) {
            v16bf af = load_fragA(cA + (wm * 64 + i * 16) * LDSS, LDSS, lane);
            c[i] = __builtin_amdgcn_wmma_f32_16x16x32_bf16(
                false, af, false, bfr, (short)0, c[i], false, false);
        }
        unsigned short* dA = swA + (cur ^ 1) * ABUF;
        unsigned short* dB = swB + (cur ^ 1) * BBUF;
        *(u32x4*)dA = ra0; *(u32x4*)(dA + 8) = ra1;
        *(u32x4*)dB = rb;
        __syncthreads();
    }
    {   // final k-step
        int cur = (nk - 1) & 1;
        const unsigned short* cA = &sA[0][0] + cur * ABUF;
        const unsigned short* cB = &sB[0][0] + cur * BBUF;
        v16bf bfr = load_fragB(cB + (wn * 16) * LDSS, LDSS, lane);
        for (int i = 0; i < 4; i++) {
            v16bf af = load_fragA(cA + (wm * 64 + i * 16) * LDSS, LDSS, lane);
            c[i] = __builtin_amdgcn_wmma_f32_16x16x32_bf16(
                false, af, false, bfr, (short)0, c[i], false, false);
        }
    }

    int b = bh / NH, h = bh % NH;
    int nloc = lane & 15;
    int moff = (lane >> 4) * 8;
    int col  = wn * 16 + nloc;                 // depth index 0..63
    for (int i = 0; i < 4; i++)
        for (int r = 0; r < 8; r++) {
            int row = m0 + wm * 64 + i * 16 + moff + r;   // sq
            ctx[((size_t)(b * SEQ + row)) * DM + h * HD + col] = f2bf(c[i][r]);
        }
}

extern "C" void kernel_launch(void* const* d_in, const int* in_sizes, int n_in,
                              void* d_out, int out_size, void* d_ws, size_t ws_size,
                              hipStream_t stream) {
    const float* q    = (const float*)d_in[0];
    const float* k    = (const float*)d_in[1];
    const float* v    = (const float*)d_in[2];
    const float* mask = (const float*)d_in[3];
    const float* Wq   = (const float*)d_in[4];
    const float* bq   = (const float*)d_in[5];
    const float* Wk   = (const float*)d_in[6];
    const float* bk   = (const float*)d_in[7];
    const float* Wv   = (const float*)d_in[8];
    const float* bv   = (const float*)d_in[9];
    const float* Wo   = (const float*)d_in[10];
    const float* bo   = (const float*)d_in[11];

    float* out_f = (float*)d_out;                       // [8192][1024]
    float* attn  = out_f + (size_t)NROWS * DM;          // [64][2048][2048] f32

    char* w = (char*)d_ws;
    size_t off = 0;
    auto alloc = [&](size_t bytes) {
        void* p = w + off;
        off += (bytes + 255) & ~(size_t)255;
        return p;
    };
    unsigned short* xq      = (unsigned short*)alloc((size_t)NROWS * DM * 2);
    unsigned short* xk      = (unsigned short*)alloc((size_t)NROWS * DM * 2);
    unsigned short* xv      = (unsigned short*)alloc((size_t)NROWS * DM * 2);
    unsigned short* Wqt     = (unsigned short*)alloc((size_t)DM * DM * 2);
    unsigned short* Wkt     = (unsigned short*)alloc((size_t)DM * DM * 2);
    unsigned short* Wvt     = (unsigned short*)alloc((size_t)DM * DM * 2);
    unsigned short* Wot     = (unsigned short*)alloc((size_t)DM * DM * 2);
    unsigned short* Qh      = (unsigned short*)alloc((size_t)NROWS * DM * 2); // [b,h,sq,64]
    unsigned short* Kh      = (unsigned short*)alloc((size_t)NROWS * DM * 2); // [b,h,sk,64]
    unsigned short* Vt      = (unsigned short*)alloc((size_t)NROWS * DM * 2); // [b,h,64,sk]
    unsigned short* ctx     = (unsigned short*)alloc((size_t)NROWS * DM * 2); // [b,sq,dm]
    unsigned short* attn_bf = (unsigned short*)alloc((size_t)B_N * NH * SEQ * SEQ * 2);

    const int nElem = NROWS * DM;
    // 1. activation converts
    f32_to_bf16_kernel<<<nElem / (256 * 8), 256, 0, stream>>>(q, xq, nElem);
    f32_to_bf16_kernel<<<nElem / (256 * 8), 256, 0, stream>>>(k, xk, nElem);
    f32_to_bf16_kernel<<<nElem / (256 * 8), 256, 0, stream>>>(v, xv, nElem);
    // 2. weight transpose+convert
    dim3 tg(DM / 32, DM / 32);
    transpose_w_kernel<<<tg, 256, 0, stream>>>(Wq, Wqt);
    transpose_w_kernel<<<tg, 256, 0, stream>>>(Wk, Wkt);
    transpose_w_kernel<<<tg, 256, 0, stream>>>(Wv, Wvt);
    transpose_w_kernel<<<tg, 256, 0, stream>>>(Wo, Wot);
    // 3. projections
    dim3 pg(NROWS / 128, DM / 128, 1);
    gemm_bf16_kernel<MODE_QK><<<pg, 256, 0, stream>>>(xq, DM, Wqt, DM, bq, nullptr, 0.f, Qh, DM);
    gemm_bf16_kernel<MODE_QK><<<pg, 256, 0, stream>>>(xk, DM, Wkt, DM, bk, nullptr, 0.f, Kh, DM);
    gemm_bf16_kernel<MODE_V ><<<pg, 256, 0, stream>>>(xv, DM, Wvt, DM, bv, nullptr, 0.f, Vt, DM);
    // 4. logits = Qh Kh^T * scale + mask * -1e9  -> attn region of d_out (f32, NT stores)
    dim3 lg(SEQ / 128, SEQ / 128, B_N * NH);
    const float scale = 0.125f;   // 1/sqrt(64)
    gemm_bf16_kernel<MODE_LOGITS><<<lg, 256, 0, stream>>>(Qh, HD, Kh, HD, nullptr, mask, scale, attn, HD);
    // 5. softmax rows in place + bf16 copy
    softmax_kernel<<<B_N * NH * SEQ, 256, 0, stream>>>(attn, attn_bf);
    // 6. ctx = attn x V  (merged heads, bf16)
    dim3 cg(SEQ / 128, 1, B_N * NH);
    ctx_gemm_kernel<<<cg, 256, 0, stream>>>(attn_bf, Vt, ctx);
    // 7. out = ctx x Wo + bo (f32)
    gemm_bf16_kernel<MODE_OUT><<<pg, 256, 0, stream>>>(ctx, DM, Wot, DM, bo, nullptr, 0.f, out_f, DM);
}
